// VoteNetHead_6923487281655
// MI455X (gfx1250) — matrix operand order repaired and checked by reading, not compiled, not run on hardware
//
#include <hip/hip_runtime.h>
#include <hip/hip_bf16.h>

// ---------------- problem constants ----------------
#define B_   8
#define M_   8192
#define C_   256
#define F_   128
#define P_   256
#define K_   32
#define CA_  256
#define NCLS_ 10
#define RAD2 0.09f
#define ROWS_TOTAL (B_ * M_)   // 65536
#define KPAD 160               // a_w1 K: 130 zero-padded to 160 (5 x 32)

// ---------------- WMMA types / helpers ----------------
typedef __attribute__((ext_vector_type(16))) __bf16 v16bf;
typedef __attribute__((ext_vector_type(8)))  float  v8f;

union Frag16  { v16bf v; unsigned short s[16]; unsigned int u[8]; };
union FragAcc { v8f   v; float f[8]; };

__device__ __forceinline__ unsigned short f2bf(float f) {
  unsigned int u = __float_as_uint(f);
  u += 0x7FFFu + ((u >> 16) & 1u);           // round-to-nearest-even
  return (unsigned short)(u >> 16);
}
__device__ __forceinline__ float bf2f(unsigned short h) {
  return __uint_as_float(((unsigned int)h) << 16);
}

// A fragment (16x32 bf16, M x K), row-major source with leading dim ld.
// ISA 7.12.2: lane l holds row m = l%16; elems 0..7 -> K = kBase + 8*(l/16)+e,
// elems 8..15 -> K = kBase + 16 + 8*(l/16) + (e-8).
__device__ __forceinline__ v16bf load_a(const unsigned short* X, int ld,
                                        int mBase, int kBase, int lane) {
  const unsigned short* r = X + (size_t)(mBase + (lane & 15)) * ld + kBase + (lane >> 4) * 8;
  Frag16 a;
#pragma unroll
  for (int e = 0; e < 8; ++e) a.s[e] = r[e];
#pragma unroll
  for (int e = 0; e < 8; ++e) a.s[8 + e] = r[16 + e];
  return a.v;
}

// B fragment (32x16 bf16, K x N) taken from weight W stored row-major (N_out x K_in):
// lane l holds column n = nBase + l%16; elems 0..7 -> K = kBase + 8*(l/16) + e,
// elems 8..15 -> K = kBase + 16 + 8*(l/16) + (e-8).
__device__ __forceinline__ v16bf load_b(const unsigned short* W, int ld,
                                        int nBase, int kBase, int lane) {
  const unsigned short* r = W + (size_t)(nBase + (lane & 15)) * ld + kBase + (lane >> 4) * 8;
  Frag16 b;
#pragma unroll
  for (int e = 0; e < 8; ++e) b.s[e] = r[e];
#pragma unroll
  for (int e = 0; e < 8; ++e) b.s[8 + e] = r[16 + e];
  return b.v;
}

__device__ __forceinline__ v8f wmma_bf16(v16bf a, v16bf b, v8f c) {
  return __builtin_amdgcn_wmma_f32_16x16x32_bf16(false, a, false, b, (short)0, c, false, false);
}

// ---------------- weight pre-conversion ----------------
__global__ void k_cvt_bf16(const float* __restrict__ src, unsigned short* __restrict__ dst, int n) {
  for (int i = blockIdx.x * blockDim.x + threadIdx.x; i < n; i += gridDim.x * blockDim.x)
    dst[i] = f2bf(src[i]);
}

__global__ void k_cvt_aw1(const float* __restrict__ src, unsigned short* __restrict__ dst) {
  const int n = CA_ * KPAD;
  for (int i = blockIdx.x * blockDim.x + threadIdx.x; i < n; i += gridDim.x * blockDim.x) {
    int r = i / KPAD, c = i - r * KPAD;
    dst[i] = (c < F_ + 2) ? f2bf(src[r * (F_ + 2) + c]) : (unsigned short)0;
  }
}

// ---------------- fused voting module ----------------
// Per block: 32 seed rows. Layer1 for both branches via WMMA (A reused),
// then offset head (2ch, VALU) and feature head (128ch, WMMA) from LDS.
__global__ __launch_bounds__(256) void k_vote(
    const float* __restrict__ seed_xyz, const float* __restrict__ seed_feats,
    const unsigned short* __restrict__ w_off1, const float* __restrict__ b_off1,
    const float* __restrict__ s_off1, const float* __restrict__ t_off1,
    const float* __restrict__ w_off2, const float* __restrict__ b_off2,
    const unsigned short* __restrict__ w_ft1, const float* __restrict__ b_ft1,
    const float* __restrict__ s_ft1, const float* __restrict__ t_ft1,
    const unsigned short* __restrict__ w_ft2, const float* __restrict__ b_ft2,
    float* __restrict__ votes, unsigned short* __restrict__ vfeat)
{
  __shared__ unsigned short Xs[32 * C_];
  __shared__ unsigned short Hs[32 * C_];
  __shared__ unsigned short Gs[32 * C_];
  const int tid  = threadIdx.x;
  const int lane = tid & 31, wave = tid >> 5;
  const int rowBase = blockIdx.x * 32;

  const float* xg = seed_feats + (size_t)rowBase * C_;
  for (int i = tid; i < 32 * C_; i += 256) Xs[i] = f2bf(xg[i]);
  __syncthreads();

  // layer1 both branches: 32x256 output each = 2x16 tiles; 4 tiles/wave
  for (int t = 0; t < 4; ++t) {
    int tile = wave * 4 + t;
    int mT = tile >> 4, nT = tile & 15;
    v8f accH = {}; v8f accG = {};
#pragma unroll
    for (int kk = 0; kk < 8; ++kk) {
      v16bf a  = load_a(Xs, C_, mT * 16, kk * 32, lane);
      v16bf bh = load_b(w_off1, C_, nT * 16, kk * 32, lane);
      v16bf bg = load_b(w_ft1,  C_, nT * 16, kk * 32, lane);
      accH = wmma_bf16(a, bh, accH);
      accG = wmma_bf16(a, bg, accG);
    }
    FragAcc h; h.v = accH; FragAcc g; g.v = accG;
    int n = nT * 16 + (lane & 15);
    float bh = b_off1[n], sh = s_off1[n], th = t_off1[n];
    float bg = b_ft1[n],  sg = s_ft1[n],  tg = t_ft1[n];
#pragma unroll
    for (int r = 0; r < 8; ++r) {
      int m = mT * 16 + r + (lane >> 4) * 8;
      Hs[m * C_ + n] = f2bf(fmaxf((h.f[r] + bh) * sh + th, 0.f));
      Gs[m * C_ + n] = f2bf(fmaxf((g.f[r] + bg) * sg + tg, 0.f));
    }
  }
  __syncthreads();

  // offset head: 32 rows x 2 channels (VALU) -> votes_xyz
  if (tid < 64) {
    int m = tid >> 1, o = tid & 1;
    const float* wrow = w_off2 + o * C_;
    float s = 0.f;
    for (int c = 0; c < C_; ++c) s += bf2f(Hs[m * C_ + c]) * wrow[c];
    s += b_off2[o];
    size_t gi = (size_t)(rowBase + m) * 2 + o;
    votes[gi] = seed_xyz[gi] + s;
  }

  // feature head: 32x128 output = 2x8 tiles; 2 tiles/wave
  for (int t = 0; t < 2; ++t) {
    int tile = wave * 2 + t;
    int mT = tile >> 3, nT = tile & 7;
    v8f acc = {};
#pragma unroll
    for (int kk = 0; kk < 8; ++kk) {
      v16bf a = load_a(Gs, C_, mT * 16, kk * 32, lane);
      v16bf b = load_b(w_ft2, C_, nT * 16, kk * 32, lane);
      acc = wmma_bf16(a, b, acc);
    }
    FragAcc d; d.v = acc;
    int n = nT * 16 + (lane & 15);
    float bb = b_ft2[n];
#pragma unroll
    for (int r = 0; r < 8; ++r) {
      int m = mT * 16 + r + (lane >> 4) * 8;
      vfeat[(size_t)(rowBase + m) * F_ + n] = f2bf(d.f[r] + bb);
    }
  }
}

// ---------------- farthest point sampling (1 block / batch) ----------------
__global__ __launch_bounds__(256) void k_fps(const float* __restrict__ votes,
                                             int* __restrict__ fps_idx,
                                             float* __restrict__ new_xyz_ws,
                                             float* __restrict__ out_newxyz)
{
  __shared__ float dist[M_];
  __shared__ float rv[256];
  __shared__ int   ri[256];
  __shared__ int   s_far;
  const int b = blockIdx.x;
  const int tid = threadIdx.x;
  const float* vx = votes + (size_t)b * M_ * 2;

  for (int j = tid; j < M_; j += 256) dist[j] = 1e10f;
  if (tid == 0) s_far = 0;
  __syncthreads();

  for (int it = 0; it < P_; ++it) {
    int far = s_far;
    float cx = vx[far * 2 + 0], cy = vx[far * 2 + 1];
    if (tid == 0) {
      fps_idx[b * P_ + it] = far;
      new_xyz_ws[(b * P_ + it) * 2 + 0] = cx;
      new_xyz_ws[(b * P_ + it) * 2 + 1] = cy;
      out_newxyz[(b * P_ + it) * 2 + 0] = cx;
      out_newxyz[(b * P_ + it) * 2 + 1] = cy;
    }
    float bestV = -1.f; int bestI = 0;
    for (int j = tid; j < M_; j += 256) {
      float dx = vx[j * 2 + 0] - cx, dy = vx[j * 2 + 1] - cy;
      float nd = fminf(dist[j], dx * dx + dy * dy);
      dist[j] = nd;
      if (nd > bestV) { bestV = nd; bestI = j; }
    }
    rv[tid] = bestV; ri[tid] = bestI;
    __syncthreads();
    for (int s = 128; s > 0; s >>= 1) {
      if (tid < s) {
        float ov = rv[tid + s]; int oi = ri[tid + s];
        if (ov > rv[tid] || (ov == rv[tid] && oi < ri[tid])) { rv[tid] = ov; ri[tid] = oi; }
      }
      __syncthreads();
    }
    if (tid == 0) s_far = ri[0];
    __syncthreads();
  }
}

// ---------------- ball query: 32 smallest-d2 (stable) per group ----------------
__global__ __launch_bounds__(256) void k_ballq(const float* __restrict__ votes,
                                               const float* __restrict__ new_xyz,
                                               int* __restrict__ gidx)
{
  __shared__ float d2[M_];
  __shared__ float rv[256];
  __shared__ int   ri[256];
  const int g = blockIdx.x;           // b*P + p
  const int b = g >> 8;
  const int tid = threadIdx.x;
  const float* vx = votes + (size_t)b * M_ * 2;
  const float qx = new_xyz[g * 2 + 0], qy = new_xyz[g * 2 + 1];

  for (int j = tid; j < M_; j += 256) {
    float dx = vx[j * 2 + 0] - qx, dy = vx[j * 2 + 1] - qy;
    d2[j] = dx * dx + dy * dy;
  }
  __syncthreads();

  for (int k = 0; k < K_; ++k) {
    float bestV = 3.4e38f; int bestI = 0;
    for (int j = tid; j < M_; j += 256) {
      float v = d2[j];
      if (v < bestV) { bestV = v; bestI = j; }
    }
    rv[tid] = bestV; ri[tid] = bestI;
    __syncthreads();
    for (int s = 128; s > 0; s >>= 1) {
      if (tid < s) {
        float ov = rv[tid + s]; int oi = ri[tid + s];
        if (ov < rv[tid] || (ov == rv[tid] && oi < ri[tid])) { rv[tid] = ov; ri[tid] = oi; }
      }
      __syncthreads();
    }
    if (tid == 0) {
      int j = ri[0];
      gidx[g * K_ + k] = (rv[0] > RAD2) ? 0 : j;
      d2[j] = 3.4e38f;                 // exclude from later picks
    }
    __syncthreads();
  }
}

// ---------------- aggregation: gather + MLP(130->256->256) + maxpool ----------------
__global__ __launch_bounds__(256) void k_agg(
    const float* __restrict__ votes, const float* __restrict__ new_xyz,
    const int* __restrict__ gidx, const unsigned short* __restrict__ vfeat,
    const unsigned short* __restrict__ a_w1, const float* __restrict__ a_b1,
    const float* __restrict__ a_s1, const float* __restrict__ a_t1,
    const unsigned short* __restrict__ a_w2, const float* __restrict__ a_b2,
    const float* __restrict__ a_s2, const float* __restrict__ a_t2,
    unsigned short* __restrict__ pooled)
{
  __shared__ unsigned short comb[K_ * KPAD];   // 32x160 bf16
  __shared__ unsigned short x1[K_ * CA_];      // 32x256 bf16
  __shared__ float          x2[K_ * CA_];      // 32x256 f32
  __shared__ int            sidx[K_];
  const int g = blockIdx.x;
  const int b = g >> 8;
  const int tid = threadIdx.x;
  const int lane = tid & 31, wave = tid >> 5;

  if (tid < K_) sidx[tid] = gidx[g * K_ + tid];
  __syncthreads();
  const float qx = new_xyz[g * 2 + 0], qy = new_xyz[g * 2 + 1];

  for (int i = tid; i < K_ * KPAD; i += 256) {
    int r = i / KPAD, c = i - r * KPAD;
    int idx = sidx[r];
    unsigned short v;
    if (c == 0)          v = f2bf(votes[((size_t)b * M_ + idx) * 2 + 0] - qx);
    else if (c == 1)     v = f2bf(votes[((size_t)b * M_ + idx) * 2 + 1] - qy);
    else if (c < F_ + 2) v = vfeat[((size_t)b * M_ + idx) * F_ + (c - 2)];
    else                 v = 0;
    comb[i] = v;
  }
  __syncthreads();

  // layer1: 32x160 @ W1^T (256x160) -> 32x256 ; 2x16 tiles, 4/wave, K=5x32
  for (int t = 0; t < 4; ++t) {
    int tile = wave * 4 + t;
    int mT = tile >> 4, nT = tile & 15;
    v8f acc = {};
#pragma unroll
    for (int kk = 0; kk < 5; ++kk) {
      v16bf a = load_a(comb, KPAD, mT * 16, kk * 32, lane);
      v16bf b = load_b(a_w1, KPAD, nT * 16, kk * 32, lane);
      acc = wmma_bf16(a, b, acc);
    }
    FragAcc d; d.v = acc;
    int n = nT * 16 + (lane & 15);
    float bb = a_b1[n], ss = a_s1[n], tt = a_t1[n];
#pragma unroll
    for (int r = 0; r < 8; ++r) {
      int m = mT * 16 + r + (lane >> 4) * 8;
      x1[m * CA_ + n] = f2bf(fmaxf((d.f[r] + bb) * ss + tt, 0.f));
    }
  }
  __syncthreads();

  // layer2: 32x256 @ W2^T (256x256) -> 32x256
  for (int t = 0; t < 4; ++t) {
    int tile = wave * 4 + t;
    int mT = tile >> 4, nT = tile & 15;
    v8f acc = {};
#pragma unroll
    for (int kk = 0; kk < 8; ++kk) {
      v16bf a = load_a(x1, CA_, mT * 16, kk * 32, lane);
      v16bf b = load_b(a_w2, CA_, nT * 16, kk * 32, lane);
      acc = wmma_bf16(a, b, acc);
    }
    FragAcc d; d.v = acc;
    int n = nT * 16 + (lane & 15);
    float bb = a_b2[n], ss = a_s2[n], tt = a_t2[n];
#pragma unroll
    for (int r = 0; r < 8; ++r) {
      int m = mT * 16 + r + (lane >> 4) * 8;
      x2[m * CA_ + n] = fmaxf((d.f[r] + bb) * ss + tt, 0.f);
    }
  }
  __syncthreads();

  // maxpool over the 32 samples
  for (int c = tid; c < CA_; c += 256) {
    float mx = x2[c];
    for (int r = 1; r < K_; ++r) mx = fmaxf(mx, x2[r * CA_ + c]);
    pooled[(size_t)g * CA_ + c] = f2bf(mx);
  }
}

// ---------------- proposal head ----------------
__global__ __launch_bounds__(256) void k_head(
    const unsigned short* __restrict__ pooled,
    const unsigned short* __restrict__ p_w1, const float* __restrict__ p_b1,
    const float* __restrict__ p_s1, const float* __restrict__ p_t1,
    const unsigned short* __restrict__ p_w2, const float* __restrict__ p_b2,
    const float* __restrict__ p_s2, const float* __restrict__ p_t2,
    const float* __restrict__ obj_w, const float* __restrict__ obj_b,
    const float* __restrict__ cls_w, const float* __restrict__ cls_b,
    float* __restrict__ out_obj, float* __restrict__ out_cls)
{
  __shared__ unsigned short y1[32 * CA_];
  __shared__ float          y2[32 * F_];
  const int tid = threadIdx.x;
  const int lane = tid & 31, wave = tid >> 5;
  const int rowBase = blockIdx.x * 32;
  const unsigned short* A0 = pooled + (size_t)rowBase * CA_;

  // layer1: 32x256 -> 32x256
  for (int t = 0; t < 4; ++t) {
    int tile = wave * 4 + t;
    int mT = tile >> 4, nT = tile & 15;
    v8f acc = {};
#pragma unroll
    for (int kk = 0; kk < 8; ++kk) {
      v16bf a = load_a(A0, CA_, mT * 16, kk * 32, lane);
      v16bf b = load_b(p_w1, CA_, nT * 16, kk * 32, lane);
      acc = wmma_bf16(a, b, acc);
    }
    FragAcc d; d.v = acc;
    int n = nT * 16 + (lane & 15);
    float bb = p_b1[n], ss = p_s1[n], tt = p_t1[n];
#pragma unroll
    for (int r = 0; r < 8; ++r) {
      int m = mT * 16 + r + (lane >> 4) * 8;
      y1[m * CA_ + n] = f2bf(fmaxf((d.f[r] + bb) * ss + tt, 0.f));
    }
  }
  __syncthreads();

  // layer2: 32x256 -> 32x128
  for (int t = 0; t < 2; ++t) {
    int tile = wave * 2 + t;
    int mT = tile >> 3, nT = tile & 7;
    v8f acc = {};
#pragma unroll
    for (int kk = 0; kk < 8; ++kk) {
      v16bf a = load_a(y1, CA_, mT * 16, kk * 32, lane);
      v16bf b = load_b(p_w2, CA_, nT * 16, kk * 32, lane);
      acc = wmma_bf16(a, b, acc);
    }
    FragAcc d; d.v = acc;
    int n = nT * 16 + (lane & 15);
    float bb = p_b2[n], ss = p_s2[n], tt = p_t2[n];
#pragma unroll
    for (int r = 0; r < 8; ++r) {
      int m = mT * 16 + r + (lane >> 4) * 8;
      y2[m * F_ + n] = fmaxf((d.f[r] + bb) * ss + tt, 0.f);
    }
  }
  __syncthreads();

  // obj (1ch) + cls (10ch) per row
  for (int t = tid; t < 32 * (NCLS_ + 1); t += 256) {
    int m = t / (NCLS_ + 1), o = t - m * (NCLS_ + 1);
    const float* w = (o == 0) ? obj_w : (cls_w + (o - 1) * F_);
    float s = 0.f;
    for (int c = 0; c < F_; ++c) s += y2[m * F_ + c] * w[c];
    if (o == 0) out_obj[rowBase + m] = s + obj_b[0];
    else        out_cls[(size_t)(rowBase + m) * NCLS_ + (o - 1)] = s + cls_b[o - 1];
  }
}

// ---------------- host launcher ----------------
extern "C" void kernel_launch(void* const* d_in, const int* in_sizes, int n_in,
                              void* d_out, int out_size, void* d_ws, size_t ws_size,
                              hipStream_t stream) {
  (void)in_sizes; (void)n_in; (void)out_size; (void)ws_size;
  const float* seed_xyz   = (const float*)d_in[0];
  const float* seed_feats = (const float*)d_in[1];
  const float* v_off_w1 = (const float*)d_in[2];
  const float* v_off_b1 = (const float*)d_in[3];
  const float* v_off_s1 = (const float*)d_in[4];
  const float* v_off_t1 = (const float*)d_in[5];
  const float* v_off_w2 = (const float*)d_in[6];
  const float* v_off_b2 = (const float*)d_in[7];
  const float* v_ft_w1  = (const float*)d_in[8];
  const float* v_ft_b1  = (const float*)d_in[9];
  const float* v_ft_s1  = (const float*)d_in[10];
  const float* v_ft_t1  = (const float*)d_in[11];
  const float* v_ft_w2  = (const float*)d_in[12];
  const float* v_ft_b2  = (const float*)d_in[13];
  const float* a_w1 = (const float*)d_in[14];
  const float* a_b1 = (const float*)d_in[15];
  const float* a_s1 = (const float*)d_in[16];
  const float* a_t1 = (const float*)d_in[17];
  const float* a_w2 = (const float*)d_in[18];
  const float* a_b2 = (const float*)d_in[19];
  const float* a_s2 = (const float*)d_in[20];
  const float* a_t2 = (const float*)d_in[21];
  const float* p_w1 = (const float*)d_in[22];
  const float* p_b1 = (const float*)d_in[23];
  const float* p_s1 = (const float*)d_in[24];
  const float* p_t1 = (const float*)d_in[25];
  const float* p_w2 = (const float*)d_in[26];
  const float* p_b2 = (const float*)d_in[27];
  const float* p_s2 = (const float*)d_in[28];
  const float* p_t2 = (const float*)d_in[29];
  const float* obj_w = (const float*)d_in[30];
  const float* obj_b = (const float*)d_in[31];
  const float* cls_w = (const float*)d_in[32];
  const float* cls_b = (const float*)d_in[33];

  // workspace carve-out (deterministic, 256B aligned)
  char* ws = (char*)d_ws;
  size_t off = 0;
  auto carve = [&](size_t bytes) -> void* {
    off = (off + 255) & ~(size_t)255;
    void* p = ws + off;
    off += bytes;
    return p;
  };
  float*          w_votes   = (float*)carve((size_t)ROWS_TOTAL * 2 * 4);
  unsigned short* w_vfeat   = (unsigned short*)carve((size_t)ROWS_TOTAL * F_ * 2);
  unsigned short* bw_off1   = (unsigned short*)carve((size_t)C_ * C_ * 2);
  unsigned short* bw_ft1    = (unsigned short*)carve((size_t)C_ * C_ * 2);
  unsigned short* bw_ft2    = (unsigned short*)carve((size_t)F_ * C_ * 2);
  unsigned short* bw_aw1    = (unsigned short*)carve((size_t)CA_ * KPAD * 2);
  unsigned short* bw_aw2    = (unsigned short*)carve((size_t)CA_ * CA_ * 2);
  unsigned short* bw_pw1    = (unsigned short*)carve((size_t)CA_ * CA_ * 2);
  unsigned short* bw_pw2    = (unsigned short*)carve((size_t)F_ * CA_ * 2);
  int*            w_fpsidx  = (int*)carve((size_t)B_ * P_ * 4);
  float*          w_newxyz  = (float*)carve((size_t)B_ * P_ * 2 * 4);
  int*            w_gidx    = (int*)carve((size_t)B_ * P_ * K_ * 4);
  unsigned short* w_pooled  = (unsigned short*)carve((size_t)B_ * P_ * CA_ * 2);

  float* out        = (float*)d_out;
  float* out_newxyz = out;                       // (8,256,2)  = 4096
  float* out_obj    = out + B_ * P_ * 2;         // (8,256)    = 2048
  float* out_cls    = out + B_ * P_ * 2 + B_ * P_;  // (8,256,10)

  // 1) weights -> bf16
  k_cvt_bf16<<<256, 256, 0, stream>>>(v_off_w1, bw_off1, C_ * C_);
  k_cvt_bf16<<<256, 256, 0, stream>>>(v_ft_w1,  bw_ft1,  C_ * C_);
  k_cvt_bf16<<<128, 256, 0, stream>>>(v_ft_w2,  bw_ft2,  F_ * C_);
  k_cvt_aw1 <<<160, 256, 0, stream>>>(a_w1,     bw_aw1);
  k_cvt_bf16<<<256, 256, 0, stream>>>(a_w2,     bw_aw2,  CA_ * CA_);
  k_cvt_bf16<<<256, 256, 0, stream>>>(p_w1,     bw_pw1,  CA_ * CA_);
  k_cvt_bf16<<<128, 256, 0, stream>>>(p_w2,     bw_pw2,  F_ * CA_);

  // 2) fused voting (WMMA)
  k_vote<<<ROWS_TOTAL / 32, 256, 0, stream>>>(
      seed_xyz, seed_feats,
      bw_off1, v_off_b1, v_off_s1, v_off_t1, v_off_w2, v_off_b2,
      bw_ft1,  v_ft_b1,  v_ft_s1,  v_ft_t1,  bw_ft2,   v_ft_b2,
      w_votes, w_vfeat);

  // 3) farthest point sampling (serial chain, 1 WGP per batch)
  k_fps<<<B_, 256, 0, stream>>>(w_votes, w_fpsidx, w_newxyz, out_newxyz);

  // 4) ball query / stable 32-NN selection
  k_ballq<<<B_ * P_, 256, 0, stream>>>(w_votes, w_newxyz, w_gidx);

  // 5) aggregation MLP + maxpool (WMMA)
  k_agg<<<B_ * P_, 256, 0, stream>>>(
      w_votes, w_newxyz, w_gidx, w_vfeat,
      bw_aw1, a_b1, a_s1, a_t1, bw_aw2, a_b2, a_s2, a_t2, w_pooled);

  // 6) proposal head (WMMA) + obj/cls
  k_head<<<(B_ * P_) / 32, 256, 0, stream>>>(
      w_pooled,
      bw_pw1, p_b1, p_s1, p_t1, bw_pw2, p_b2, p_s2, p_t2,
      obj_w, obj_b, cls_w, cls_b, out_obj, out_cls);
}
